// SegmentalLM_58256936403340
// MI455X (gfx1250) — matrix operand order repaired and checked by reading, not compiled, not run on hardware
//
#include <hip/hip_runtime.h>
#include <cstddef>

typedef __attribute__((ext_vector_type(16))) _Float16 v16h;
typedef __attribute__((ext_vector_type(8)))  _Float16 v8h;
typedef __attribute__((ext_vector_type(8)))  float    v8f;

#define LOGINF 1000000.0f

// ---------------------------------------------------------------------------
// WMMA helpers (CDNA5 gfx1250, wave32)
// ---------------------------------------------------------------------------
__device__ __forceinline__ v8f wmma_f16(v16h a, v16h b, v8f c) {
  // (neg_a, A, neg_b, B, c_mod, C, reuse_a, reuse_b)
  return __builtin_amdgcn_wmma_f32_16x16x32_f16(false, a, false, b, (short)0, c,
                                                false, false);
}

// A fragment, 16x32 f16 (M x K), row-major source with row stride `ld` halves.
// Layout: lanes 0-15 hold K 0..15, lanes 16-31 hold K 16..31
// (contiguous, matching B).  Row = lane & 15.
__device__ __forceinline__ v16h load_a_frag(const _Float16* base, int ld, int lane) {
  const _Float16* p = base + (size_t)(lane & 15) * ld + ((lane >> 4) << 4);
  v8h lo = *(const v8h*)(p);
  v8h hi = *(const v8h*)(p + 8);
  return __builtin_shufflevector(lo, hi, 0, 1, 2, 3, 4, 5, 6, 7,
                                 8, 9, 10, 11, 12, 13, 14, 15);
}

// B fragment, 32x16 f16 (K x N), from BT stored [N, K] row-major (ldk halves).
// Layout (contiguous, same order as A): column = lane&15; lanes 0-15 hold K 0..15, lanes 16-31 K 16..31
// (each is one contiguous 32-byte run).
__device__ __forceinline__ v16h load_b_frag(const _Float16* bt, int ldk,
                                            int col0, int kbase, int lane) {
  const _Float16* p = bt + (size_t)(col0 + (lane & 15)) * ldk + kbase +
                      ((lane >> 4) << 4);
  return *(const v16h*)p;
}

__device__ __forceinline__ float sigf(float x) { return 1.0f / (1.0f + expf(-x)); }

// ---------------------------------------------------------------------------
// Small utility kernels
// ---------------------------------------------------------------------------
__global__ void k_cvt_f16(const float* __restrict__ in, _Float16* __restrict__ out,
                          int n) {
  int i = blockIdx.x * 256 + threadIdx.x;
  if (i < n) out[i] = (_Float16)in[i];
}

// in [K,N] f32 -> out [N,K] f16   (so B columns become contiguous K runs)
__global__ void k_tcvt(const float* __restrict__ in, _Float16* __restrict__ out,
                       int K, int N) {
  int i = blockIdx.x * 256 + threadIdx.x;
  if (i >= K * N) return;
  int n = i / K, k = i % K;
  out[i] = (_Float16)in[(size_t)k * N + n];
}

// embPad[(t*8+b)*256 + d] = embedding[x[b,t]][d]  (t<128; tail rows pre-zeroed)
__global__ void k_gather(const int* __restrict__ x, const float* __restrict__ emb,
                         _Float16* __restrict__ embPad) {
  int i = blockIdx.x * 256 + threadIdx.x;          // 128*8*256
  int d = i & 255;
  int rb = i >> 8;
  int t = rb >> 3, b = rb & 7;
  int tok = x[b * 128 + t];
  embPad[(size_t)(t * 8 + b) * 256 + d] = (_Float16)emb[(size_t)tok * 256 + d];
}

// ---------------------------------------------------------------------------
// Generic WMMA GEMM: C[rows,N] = A[rows,256] @ B[256,N] (+bias)(tanh)(f16/f32)
// A: f16 row-major [rows,256]; BT: f16 [N,256]. block = 256 threads = 8 waves.
// grid = (rows/16, N/128); wave w -> cols blockIdx.y*128 + w*16.
// ---------------------------------------------------------------------------
template <int HASBIAS, int TANH, int F16OUT>
__global__ void __launch_bounds__(256) k_gemm(const _Float16* __restrict__ A,
                                              const _Float16* __restrict__ BT,
                                              const float* __restrict__ bias,
                                              float* __restrict__ Cf,
                                              _Float16* __restrict__ Ch, int N) {
  int lane = threadIdx.x & 31, w = threadIdx.x >> 5;
  int row0 = blockIdx.x * 16;
  int col0 = blockIdx.y * 128 + w * 16;
  if (col0 >= N) return;

  v8f c = {0.f, 0.f, 0.f, 0.f, 0.f, 0.f, 0.f, 0.f};
#pragma unroll
  for (int k = 0; k < 8; k++) {
    v16h a = load_a_frag(A + (size_t)row0 * 256 + k * 32, 256, lane);
    v16h b = load_b_frag(BT, 256, col0, k * 32, lane);
    c = wmma_f16(a, b, c);
  }

  int nc = lane & 15, rh = lane >> 4;
  int col = col0 + nc;
  float bv = HASBIAS ? bias[col] : 0.f;
#pragma unroll
  for (int i = 0; i < 8; i++) {
    int row = row0 + i + 8 * rh;
    float v = c[i] + bv;
    if (TANH) v = tanhf(v);
    if (F16OUT)
      Ch[(size_t)row * N + col] = (_Float16)v;
    else
      Cf[(size_t)row * N + col] = v;
  }
}

// ---------------------------------------------------------------------------
// Encoder LSTM: persistent single workgroup, 128 sequential steps.
// Batch 8 padded to 16 rows; 8 waves cover all 1024 gate columns:
// wave w owns hidden cols [w*32, w*32+32) across the 4 gate blocks.
// h tile is double-buffered in LDS (one barrier per step).  An opaque integer
// offset (laundered through inline asm) blocks hoisting of the W_hh
// B-fragments (prevents scratch spills) while preserving the kernarg pointer
// provenance, so the loads stay global_* (LOADcnt only) instead of flat_*
// (LOADcnt+DScnt, which would falsely couple with the LDS h-tile reads).
// ---------------------------------------------------------------------------
__global__ void __launch_bounds__(256) k_enc_lstm(const float* __restrict__ Xg,
                                                  const _Float16* __restrict__ WhhT,
                                                  _Float16* __restrict__ hsE) {
  __shared__ _Float16 hS[2][16 * 256];  // double-buffered h tile (16 KB)
  int tid = threadIdx.x, lane = tid & 31, w = tid >> 5;
  for (int i = tid; i < 16 * 256; i += 256) {
    hS[0][i] = (_Float16)0.f;  // rows 8..15 stay zero forever (batch padding)
    hS[1][i] = (_Float16)0.f;
  }
  float cReg[2][8];
#pragma unroll
  for (int cg = 0; cg < 2; cg++)
#pragma unroll
    for (int i = 0; i < 8; i++) cReg[cg][i] = 0.f;
  __syncthreads();

  int lz = 0;  // opaque zero: re-laundered every step
  for (int t = 0; t < 128; t++) {
    asm volatile("" : "+s"(lz));
    const _Float16* wp = WhhT + lz;
    const _Float16* hRead = hS[t & 1];
    _Float16* hWrite = hS[(t & 1) ^ 1];

    v8f acc[2][4];
#pragma unroll
    for (int cg = 0; cg < 2; cg++)
#pragma unroll
      for (int g = 0; g < 4; g++)
        acc[cg][g] = (v8f){0.f, 0.f, 0.f, 0.f, 0.f, 0.f, 0.f, 0.f};

#pragma unroll
    for (int k = 0; k < 8; k++) {
      v16h a = load_a_frag(hRead + k * 32, 256, lane);
#pragma unroll
      for (int cg = 0; cg < 2; cg++)
#pragma unroll
        for (int g = 0; g < 4; g++) {
          v16h b = load_b_frag(wp, 256, g * 256 + w * 32 + cg * 16, k * 32, lane);
          acc[cg][g] = wmma_f16(a, b, acc[cg][g]);
        }
    }

    int nc = lane & 15, rh = lane >> 4;
#pragma unroll
    for (int cg = 0; cg < 2; cg++) {
      int col = w * 32 + cg * 16 + nc;
#pragma unroll
      for (int i = 0; i < 8; i++) {
        int row = i + 8 * rh;
        if (row < 8) {  // real batch rows; rows 8..15 are zero padding
          const float* xr = Xg + (size_t)(t * 8 + row) * 1024;
          float gi = acc[cg][0][i] + xr[col];
          float gf = acc[cg][1][i] + xr[col + 256];
          float gg = acc[cg][2][i] + xr[col + 512];
          float go = acc[cg][3][i] + xr[col + 768];
          float cN = sigf(gf) * cReg[cg][i] + sigf(gi) * tanhf(gg);
          cReg[cg][i] = cN;
          float h = sigf(go) * tanhf(cN);
          hWrite[row * 256 + col] = (_Float16)h;
          hsE[(size_t)(t * 8 + row) * 256 + col] = (_Float16)(0.5f * h);
        }
      }
    }
    __syncthreads();
  }
}

// ---------------------------------------------------------------------------
// Decoder LSTM step: all 126 starts x 8 batch = 1008 rows, grid = 63 tiles.
// Xpre row for state row q: ((q>>3)+addJ)*8 + (q&7).  Ping-pong h buffers.
// ---------------------------------------------------------------------------
__global__ void __launch_bounds__(256) k_dec_step(const float* __restrict__ Xpre,
                                                  int addJ,
                                                  const _Float16* __restrict__ WhhT,
                                                  const _Float16* __restrict__ hIn,
                                                  _Float16* __restrict__ hOut,
                                                  float* __restrict__ cState,
                                                  _Float16* __restrict__ hsAll,
                                                  int s) {
  int lane = threadIdx.x & 31, w = threadIdx.x >> 5;
  int row0 = blockIdx.x * 16;

  v8f acc[2][4];
#pragma unroll
  for (int cg = 0; cg < 2; cg++)
#pragma unroll
    for (int g = 0; g < 4; g++)
      acc[cg][g] = (v8f){0.f, 0.f, 0.f, 0.f, 0.f, 0.f, 0.f, 0.f};

#pragma unroll
  for (int k = 0; k < 8; k++) {
    v16h a = load_a_frag(hIn + (size_t)row0 * 256 + k * 32, 256, lane);
#pragma unroll
    for (int cg = 0; cg < 2; cg++)
#pragma unroll
      for (int g = 0; g < 4; g++) {
        v16h b = load_b_frag(WhhT, 256, g * 256 + w * 32 + cg * 16, k * 32, lane);
        acc[cg][g] = wmma_f16(a, b, acc[cg][g]);
      }
  }

  int nc = lane & 15, rh = lane >> 4;
#pragma unroll
  for (int cg = 0; cg < 2; cg++) {
    int col = w * 32 + cg * 16 + nc;
#pragma unroll
    for (int i = 0; i < 8; i++) {
      int q = row0 + i + 8 * rh;  // rows are exactly 1008 = 63*16, all valid
      int preRow = ((q >> 3) + addJ) * 8 + (q & 7);
      const float* xr = Xpre + (size_t)preRow * 1024;
      float gi = acc[cg][0][i] + xr[col];
      float gf = acc[cg][1][i] + xr[col + 256];
      float gg = acc[cg][2][i] + xr[col + 512];
      float go = acc[cg][3][i] + xr[col + 768];
      float cOld = cState[(size_t)q * 256 + col];
      float cN = sigf(gf) * cOld + sigf(gi) * tanhf(gg);
      cState[(size_t)q * 256 + col] = cN;
      float h = sigf(go) * tanhf(cN);
      hOut[(size_t)q * 256 + col] = (_Float16)h;
      size_t hsRow = (size_t)((q >> 3) * 5 + s) * 8 + (q & 7);
      hsAll[hsRow * 256 + col] = (_Float16)h;
    }
  }
}

// ---------------------------------------------------------------------------
// Dominant kernel: logits = hp[5040,256] @ embF16^T[256,32000], fused online
// logsumexp per row (logits never touch memory). grid = 315 row tiles,
// 8 waves x 16 vocab cols per iteration, 250 iterations of 128 cols.
// Per-lane online (m,s): ONE expf per element per tile, no shuffles in the
// hot loop; cross-lane and cross-wave merges happen once at the end.
// ---------------------------------------------------------------------------
__global__ void __launch_bounds__(256) k_logits_lse(const _Float16* __restrict__ hp,
                                                    const _Float16* __restrict__ embF16,
                                                    float* __restrict__ lse,
                                                    int rows) {
  __shared__ float mS[8][16];
  __shared__ float sS[8][16];
  int lane = threadIdx.x & 31, w = threadIdx.x >> 5;
  int row0 = blockIdx.x * 16;

  v16h a[8];
#pragma unroll
  for (int k = 0; k < 8; k++)
    a[k] = load_a_frag(hp + (size_t)row0 * 256 + k * 32, 256, lane);

  float m[8], sum[8];
#pragma unroll
  for (int i = 0; i < 8; i++) { m[i] = -1e30f; sum[i] = 0.f; }

  for (int nb = 0; nb < 250; nb++) {
    int v0 = nb * 128 + w * 16;
    v8f c = {0.f, 0.f, 0.f, 0.f, 0.f, 0.f, 0.f, 0.f};
#pragma unroll
    for (int k = 0; k < 8; k++) {
      v16h b = load_b_frag(embF16, 256, v0, k * 32, lane);
      c = wmma_f16(a[k], b, c);
    }
    if (nb + 1 < 250)  // global_prefetch_b8 next B tile (L2-resident stream)
      __builtin_prefetch(embF16 + (size_t)(v0 + 128 + (lane & 15)) * 256, 0, 3);

    // per-lane online logsumexp update: one expf per element
#pragma unroll
    for (int i = 0; i < 8; i++) {
      float cv = c[i];
      float d = cv - m[i];
      float ed = expf(-fabsf(d));
      bool up = d > 0.f;
      sum[i] = up ? (sum[i] * ed + 1.f) : (sum[i] + ed);
      m[i] = up ? cv : m[i];
    }
  }

  // merge across the 16 lanes of each half-wave (rows i / i+8)
#pragma unroll
  for (int dlt = 1; dlt < 16; dlt <<= 1) {
#pragma unroll
    for (int i = 0; i < 8; i++) {
      float mo = __shfl_xor(m[i], dlt, 32);
      float so = __shfl_xor(sum[i], dlt, 32);
      float nm = fmaxf(m[i], mo);
      sum[i] = sum[i] * expf(m[i] - nm) + so * expf(mo - nm);
      m[i] = nm;
    }
  }

  if (lane == 0)
#pragma unroll
    for (int i = 0; i < 8; i++) { mS[w][i] = m[i]; sS[w][i] = sum[i]; }
  if (lane == 16)
#pragma unroll
    for (int i = 0; i < 8; i++) { mS[w][8 + i] = m[i]; sS[w][8 + i] = sum[i]; }
  __syncthreads();

  if (threadIdx.x < 16) {
    float M = -1e30f, S = 0.f;
#pragma unroll
    for (int ww = 0; ww < 8; ww++) {
      float mm = mS[ww][threadIdx.x], ss = sS[ww][threadIdx.x];
      float nm = fmaxf(M, mm);
      S = S * expf(M - nm) + ss * expf(mm - nm);
      M = nm;
    }
    int r = row0 + threadIdx.x;
    if (r < rows) lse[r] = M + logf(S);
  }
}

// ---------------------------------------------------------------------------
// Recover the only two logits actually used per row: target token and EOS.
// One wave per row; 256-length f32 dot.
// ---------------------------------------------------------------------------
__global__ void __launch_bounds__(256) k_dots(const _Float16* __restrict__ hp,
                                              const _Float16* __restrict__ embF16,
                                              const int* __restrict__ x,
                                              float* __restrict__ tgtL,
                                              float* __restrict__ eosL) {
  int w = threadIdx.x >> 5, lane = threadIdx.x & 31;
  int r = blockIdx.x * 8 + w;
  if (r >= 5040) return;
  int j = r / 40 + 1, s = (r >> 3) % 5, b = r & 7;
  int tpos = j + s;
  int tok = (s < 4 && tpos < 128) ? x[b * 128 + tpos] : 0;
  const _Float16* h = hp + (size_t)r * 256;
  const _Float16* et = embF16 + (size_t)tok * 256;
  const _Float16* ee = embF16 + (size_t)1 * 256;  // EOS = 1
  float d1 = 0.f, d2 = 0.f;
  for (int i = lane; i < 256; i += 32) {
    float hv = (float)h[i];
    d1 += hv * (float)et[i];
    d2 += hv * (float)ee[i];
  }
#pragma unroll
  for (int d = 16; d >= 1; d >>= 1) {
    d1 += __shfl_xor(d1, d, 32);
    d2 += __shfl_xor(d2, d, 32);
  }
  if (lane == 0) { tgtL[r] = d1; eosL[r] = d2; }
}

// ---------------------------------------------------------------------------
// Build logpy[127][4][8]: cumsum of target log-probs + single-char penalties
// + EOS log-prob + 1.0, with validity masking.
// ---------------------------------------------------------------------------
__global__ void k_logpy(const int* __restrict__ x, const float* __restrict__ tgtL,
                        const float* __restrict__ eosL, const float* __restrict__ lse,
                        float* __restrict__ logpy) {
  int tid = blockIdx.x * 256 + threadIdx.x;
  if (tid >= 127 * 8) return;
  int j = tid / 8, b = tid % 8;
  if (j == 0) {
    for (int k = 0; k < 4; k++)
      logpy[(size_t)(0 * 4 + k) * 8 + b] = (k == 0) ? 0.f : -LOGINF;
    return;
  }
  auto isSingle = [&](int t) -> float {
    if (t >= 128) return 0.f;
    int tk = x[b * 128 + t];
    return (tk == 2 || tk == 3 || tk == 4) ? -LOGINF : 0.f;
  };
  float issw0 = isSingle(j);
  float cum = 0.f;
  for (int k = 0; k < 4; k++) {
    int r = (j - 1) * 40 + k * 8 + b;
    float tlp = tgtL[r] - lse[r];
    float add = (k == 0) ? 0.f : isSingle(j + k);
    cum += tlp + add;
    float c2 = cum + ((k >= 1) ? issw0 : 0.f);
    int rE = (j - 1) * 40 + (k + 1) * 8 + b;
    float elp = eosL[rE] - lse[rE];
    float val = c2 + elp + 1.0f;
    bool valid = (k <= 126 - j);
    logpy[(size_t)(j * 4 + k) * 8 + b] = valid ? val : -LOGINF;
  }
}

// ---------------------------------------------------------------------------
// Final DP: single wave32; lane = b*4 + k.  126 sequential logsumexp steps.
// ---------------------------------------------------------------------------
__global__ void k_dp(const float* __restrict__ logpy, const int* __restrict__ lengths,
                     float* __restrict__ out) {
  __shared__ float alpha[127][8];
  int l = threadIdx.x;
  int b = l >> 2, k = l & 3;
  float buf = (k == 3) ? 0.f : -LOGINF;
  if (l < 8) alpha[0][l] = 0.f;
  for (int mstep = 1; mstep <= 126; mstep++) {
    float d = (mstep - k >= 1) ? logpy[(size_t)((mstep - k) * 4 + k) * 8 + b]
                               : -LOGINF;
    float rev = __shfl_xor(buf, 3, 32);  // buf[3-k] within the 4-lane group
    float c = rev + d;
    float mx = fmaxf(c, __shfl_xor(c, 1, 32));
    mx = fmaxf(mx, __shfl_xor(mx, 2, 32));
    float e = expf(c - mx);
    e += __shfl_xor(e, 1, 32);
    e += __shfl_xor(e, 2, 32);
    float a = mx + logf(e);
    float shifted = __shfl_down(buf, 1, 32);
    buf = (k == 3) ? a : shifted;
    if (k == 0) alpha[mstep][b] = a;
  }
  __syncthreads();
  if (l == 0) {
    float tot = 0.f;
    int sl = 0;
    for (int bb = 0; bb < 8; bb++) {
      int Lb = lengths[bb];
      sl += Lb;
      tot += -alpha[Lb - 2][bb];
    }
    out[0] = tot / (float)(sl - 16);
  }
}

// ---------------------------------------------------------------------------
// Host launcher
// ---------------------------------------------------------------------------
extern "C" void kernel_launch(void* const* d_in, const int* in_sizes, int n_in,
                              void* d_out, int out_size, void* d_ws, size_t ws_size,
                              hipStream_t stream) {
  const int*   x        = (const int*)d_in[0];
  const int*   lengths  = (const int*)d_in[1];
  const float* embedding= (const float*)d_in[2];
  const float* W_ih_e   = (const float*)d_in[3];
  const float* W_hh_e   = (const float*)d_in[4];
  const float* b_e      = (const float*)d_in[5];
  const float* W_start  = (const float*)d_in[6];
  const float* W_ih_d   = (const float*)d_in[7];
  const float* W_hh_d   = (const float*)d_in[8];
  const float* b_d      = (const float*)d_in[9];
  const float* W_proj   = (const float*)d_in[10];
  float* out = (float*)d_out;

  char* W = (char*)d_ws;
  size_t o = 0;
  auto alloc = [&](size_t bytes) -> char* {
    char* p = W + o;
    o += (bytes + 255) & ~(size_t)255;
    return p;
  };
  _Float16* embF16 = (_Float16*)alloc((size_t)32000 * 256 * 2);
  _Float16* WihTe  = (_Float16*)alloc((size_t)1024 * 256 * 2);
  _Float16* WhhTe  = (_Float16*)alloc((size_t)1024 * 256 * 2);
  _Float16* WihTd  = (_Float16*)alloc((size_t)1024 * 256 * 2);
  _Float16* WhhTd  = (_Float16*)alloc((size_t)1024 * 256 * 2);
  _Float16* WstartT= (_Float16*)alloc((size_t)256 * 256 * 2);
  _Float16* WprojT = (_Float16*)alloc((size_t)256 * 256 * 2);
  _Float16* embPad = (_Float16*)alloc((size_t)1056 * 256 * 2);
  float*    XgE    = (float*)   alloc((size_t)1024 * 1024 * 4);
  _Float16* hsE    = (_Float16*)alloc((size_t)1024 * 256 * 2);
  _Float16* segdec = (_Float16*)alloc((size_t)1024 * 256 * 2);
  float*    EgD    = (float*)   alloc((size_t)1056 * 1024 * 4);
  float*    SgD    = (float*)   alloc((size_t)1008 * 1024 * 4);
  float*    cD     = (float*)   alloc((size_t)1008 * 256 * 4);
  _Float16* hDa    = (_Float16*)alloc((size_t)1008 * 256 * 2);
  _Float16* hDb    = (_Float16*)alloc((size_t)1008 * 256 * 2);
  _Float16* hsAll  = (_Float16*)alloc((size_t)5040 * 256 * 2);
  _Float16* hp     = (_Float16*)alloc((size_t)5040 * 256 * 2);
  float*    lse    = (float*)   alloc((size_t)5040 * 4);
  float*    tgtL   = (float*)   alloc((size_t)5040 * 4);
  float*    eosL   = (float*)   alloc((size_t)5040 * 4);
  float*    logpy  = (float*)   alloc((size_t)127 * 4 * 8 * 4);
  (void)ws_size; (void)n_in; (void)in_sizes; (void)out_size;

  // --- conversions / transposes ---
  k_cvt_f16<<<(32000 * 256) / 256, 256, 0, stream>>>(embedding, embF16, 32000 * 256);
  k_tcvt<<<1024, 256, 0, stream>>>(W_ih_e, WihTe, 256, 1024);
  k_tcvt<<<1024, 256, 0, stream>>>(W_hh_e, WhhTe, 256, 1024);
  k_tcvt<<<1024, 256, 0, stream>>>(W_ih_d, WihTd, 256, 1024);
  k_tcvt<<<1024, 256, 0, stream>>>(W_hh_d, WhhTd, 256, 1024);
  k_tcvt<<<256, 256, 0, stream>>>(W_start, WstartT, 256, 256);
  k_tcvt<<<256, 256, 0, stream>>>(W_proj, WprojT, 256, 256);

  hipMemsetAsync(embPad, 0, (size_t)1056 * 256 * 2, stream);
  hipMemsetAsync(cD, 0, (size_t)1008 * 256 * 4, stream);
  hipMemsetAsync(hDa, 0, (size_t)1008 * 256 * 2, stream);

  k_gather<<<1024, 256, 0, stream>>>(x, embedding, embPad);

  // --- encoder: input-gate GEMM then sequential recurrent WMMA kernel ---
  k_gemm<1, 0, 0><<<dim3(64, 8), 256, 0, stream>>>(embPad, WihTe, b_e, XgE, nullptr, 1024);
  k_enc_lstm<<<1, 256, 0, stream>>>(XgE, WhhTe, hsE);

  // seg_dec_hiddens = tanh((0.5*h) @ W_start)  -> f16
  k_gemm<0, 1, 1><<<dim3(64, 2), 256, 0, stream>>>(hsE, WstartT, nullptr, nullptr, segdec, 256);

  // --- decoder input-gate precompute ---
  k_gemm<1, 0, 0><<<dim3(66, 8), 256, 0, stream>>>(embPad, WihTd, b_d, EgD, nullptr, 1024);
  k_gemm<1, 0, 0><<<dim3(63, 8), 256, 0, stream>>>(segdec, WihTd, b_d, SgD, nullptr, 1024);

  // --- decoder: 5 sequential steps over all 126 starts x 8 batch ---
  for (int s = 0; s < 5; s++) {
    const float* pre = (s == 0) ? SgD : EgD;
    int addJ = s;  // s==0 -> preRow == q (SgD); s>=1 -> Eg row (j+s-1)*8+b
    _Float16* hIn = (s % 2 == 0) ? hDa : hDb;
    _Float16* hOut = (s % 2 == 0) ? hDb : hDa;
    k_dec_step<<<63, 256, 0, stream>>>(pre, addJ, WhhTd, hIn, hOut, cD, hsAll, s);
  }

  // hp = h @ W_proj  (f16)
  k_gemm<0, 0, 1><<<dim3(315, 2), 256, 0, stream>>>(hsAll, WprojT, nullptr, nullptr, hp, 256);

  // --- dominant: logits GEMM + fused online logsumexp over V=32000 ---
  k_logits_lse<<<315, 256, 0, stream>>>(hp, embF16, lse, 5040);

  // recover target/EOS logits, assemble logpy, run DP
  k_dots<<<630, 256, 0, stream>>>(hp, embF16, x, tgtL, eosL);
  k_logpy<<<4, 256, 0, stream>>>(x, tgtL, eosL, lse, logpy);
  k_dp<<<1, 32, 0, stream>>>(logpy, lengths, out);
}
